// RelativeGlobalAttention_64072322122494
// MI455X (gfx1250) — compile-verified, matrix operand order checked by reading, not run on hardware
//
#include <hip/hip_runtime.h>
#include <math.h>

#define DEV __device__ __forceinline__

typedef __attribute__((ext_vector_type(16))) __bf16 bf16x16;
typedef __attribute__((ext_vector_type(8)))  __bf16 bf16x8;
typedef __attribute__((ext_vector_type(8)))  float  f32x8;
typedef __attribute__((ext_vector_type(4)))  float  f32x4;
typedef unsigned short u16;
typedef unsigned int   u32;
typedef unsigned long long u64;
typedef __attribute__((ext_vector_type(4))) u32 u32x4;
typedef __attribute__((ext_vector_type(8))) int i32x8;
typedef __attribute__((ext_vector_type(4))) int i32x4;

constexpr int Bn = 4, Ln = 2048, Dn = 512, Hn = 8, DHn = 64, Mseq = 2048;

#if __has_builtin(__builtin_amdgcn_tensor_load_to_lds)
#define HAVE_TDM 1
#else
#define HAVE_TDM 0
#endif

DEV u16 f2bf(float f) {
  union { float f; u32 u; } v; v.f = f;
  u32 r = v.u + 0x7FFFu + ((v.u >> 16) & 1u);   // RNE
  return (u16)(r >> 16);
}

// Pack two f32 -> packed bf16x2 in a u32 (hardware cvt_pk if available).
DEV u32 pk2(float a, float b) {
#if __has_builtin(__builtin_amdgcn_cvt_pk_bf16_f32)
  auto p = __builtin_amdgcn_cvt_pk_bf16_f32(a, b);
  return __builtin_bit_cast(u32, p);
#else
  union { float f; u32 u; } x, y; x.f = a; y.f = b;
  u32 ra = (x.u + 0x7FFFu + ((x.u >> 16) & 1u)) >> 16;
  u32 rb = (y.u + 0x7FFFu + ((y.u >> 16) & 1u)) & 0xFFFF0000u;
  return ra | rb;
#endif
}

// Async DMA copy of 16 bytes global -> LDS (gfx1250 ASYNCcnt path).
DEV void async_b128(const void* g, void* lds) {
  u32 loff = (u32)(size_t)lds;   // low 32 bits of generic shared ptr = LDS offset
  asm volatile("global_load_async_to_lds_b128 %0, %1, off"
               :: "v"(loff), "v"(g) : "memory");
}
DEV void async_wait() { asm volatile("s_wait_asynccnt 0x0" ::: "memory"); }

#if HAVE_TDM
// TDM: DMA a 2-D tile (16-bit elements) from global memory into LDS.
// D# per CDNA5 ISA ch.8: group0 = {flags/count, lds_addr, global_addr, type},
// group1 = {data_size, tensor dims, tile dims, stride}; groups 2/3 unused (2-D).
// 6-arg builtin form (amdgpu-toolchain / therock headers).
DEV void tdm_load_tile_2d_b16(u32 lds_off, const void* gptr,
                              u32 dim0, u32 dim1, u32 stride0) {
  u64 ga = (u64)(size_t)gptr;
  u32x4 g0;
  g0[0] = 1u;                                   // count=1, user mode
  g0[1] = lds_off;                              // lds_addr (bytes)
  g0[2] = (u32)ga;                              // global_addr[31:0]
  g0[3] = ((u32)(ga >> 32) & 0x01FFFFFFu) | (2u << 30);  // addr[56:32] | type=2
  i32x8 g1;
  g1[0] = (int)(1u << 16);                      // data_size=1 -> 2 bytes
  g1[1] = (int)((dim0 & 0xFFFFu) << 16);        // tensor_dim0[15:0]
  g1[2] = (int)((dim0 >> 16) | ((dim1 & 0xFFFFu) << 16));
  g1[3] = (int)((dim1 >> 16) | ((dim0 & 0xFFFFu) << 16)); // tile_dim0 = dim0
  g1[4] = (int)(dim1 & 0xFFFFu);                // tile_dim1 = dim1, tile_dim2=0
  g1[5] = (int)stride0;                         // tensor_dim0_stride[31:0]
  g1[6] = 0;
  g1[7] = 0;
  i32x4 gz = {0, 0, 0, 0};
  i32x8 gz8 = {0, 0, 0, 0, 0, 0, 0, 0};
  __builtin_amdgcn_tensor_load_to_lds(g0, g1, gz, gz, gz8, 0);
}
#endif

// Load a 16x32 bf16 fragment from an LDS tile laid out [row][K], K-contiguous.
// CDNA5 16-bit WMMA layout: lanes 0-15 K 0..7/16..23, lanes 16-31 K 8..15/24..31.
DEV bf16x16 load_frag(const u16* base, int stride) {
  int lane = threadIdx.x & 31;
  int r = lane & 15;
  int kb = (lane & 16) >> 1;            // 0 or 8
  const u16* p = base + r * stride + kb;
  bf16x8 lo = *(const bf16x8*)(p);      // K = kb + 0..7
  bf16x8 hi = *(const bf16x8*)(p + 16); // K = kb + 16..23
  return __builtin_shufflevector(lo, hi, 0,1,2,3,4,5,6,7,8,9,10,11,12,13,14,15);
}

DEV f32x8 wmma_bf16(bf16x16 a, bf16x16 b, f32x8 c) {
  return __builtin_amdgcn_wmma_f32_16x16x32_bf16(false, a, false, b, (short)0, c,
                                                 false, false);
}

// ---------------- Kernel 1: projection GEMM  Y = X @ W + b  -> bf16 head-major
__global__ __launch_bounds__(128)
void proj_gemm_kernel(const float* __restrict__ X, const float* __restrict__ W,
                      const float* __restrict__ bias, u16* __restrict__ Yh) {
  __shared__ __align__(16) u16 As[64][32];   // [M][K]
  __shared__ __align__(16) u16 Bs[64][32];   // [N][K] (transposed on store)
  const int t = threadIdx.x, lane = t & 31, w = t >> 5;
  const int n0 = blockIdx.x * 64;
  const int m0 = blockIdx.y * 64;
  f32x8 acc[4] = {};
  for (int k0 = 0; k0 < Dn; k0 += 32) {
    #pragma unroll
    for (int p = 0; p < 4; ++p) {            // A: 64x32 f32 -> packed bf16
      int row = p * 16 + (t >> 3);
      int col = (t & 7) * 4;
      f32x4 v = *(const f32x4*)(X + (size_t)(m0 + row) * Dn + k0 + col);
      uint2 pv = { pk2(v[0], v[1]), pk2(v[2], v[3]) };
      *(uint2*)&As[row][col] = pv;
    }
    {                                        // B: column-per-thread, packed store
      int n  = t & 63;
      int kg = (t >> 6) * 16;
      #pragma unroll
      for (int kk = 0; kk < 16; kk += 2) {
        float a = W[(size_t)(k0 + kg + kk)     * Dn + n0 + n];
        float b = W[(size_t)(k0 + kg + kk + 1) * Dn + n0 + n];
        *(u32*)&Bs[n][kg + kk] = pk2(a, b);
      }
    }
    if (k0 + 32 < Dn) {
      __builtin_prefetch(X + (size_t)(m0 + (t >> 1)) * Dn + k0 + 32, 0, 0);
      __builtin_prefetch(W + (size_t)(k0 + 32 + (t >> 2)) * Dn + n0, 0, 0);
    }
    __syncthreads();
    bf16x16 a = load_frag(&As[w * 16][0], 32);
    #pragma unroll
    for (int ns = 0; ns < 4; ++ns)
      acc[ns] = wmma_bf16(a, load_frag(&Bs[ns * 16][0], 32), acc[ns]);
    __syncthreads();
  }
  const int hi8 = (lane & 16) >> 1;
  #pragma unroll
  for (int ns = 0; ns < 4; ++ns) {
    int n = n0 + ns * 16 + (lane & 15);
    int h = n >> 6, dh = n & 63;
    float bv = bias[n];
    #pragma unroll
    for (int e = 0; e < 8; ++e) {
      int m = m0 + w * 16 + e + hi8;
      int b = m >> 11, l = m & (Ln - 1);
      Yh[(((size_t)(b * Hn + h) * Ln) + l) * DHn + dh] = f2bf(acc[ns][e] + bv);
    }
  }
}

// ---------------- Kernel 2: logits = (Q Kt + Srel) / sqrt(dh)
// Srel[i,j] = dot(q_i, E[M-1-(i-j)]) for j<=i, else 0 (skew trick solved).
__global__ __launch_bounds__(128)
void logits_kernel(const u16* __restrict__ Qh, const u16* __restrict__ Kh,
                   const float* __restrict__ E, float* __restrict__ attn) {
  __shared__ __align__(16) u16 Qs[16][64];
  __shared__ __align__(16) u16 Ks[64][64];       // rows = local j, cols = dh
  __shared__ __align__(16) u16 Es[4][32][64];    // per-wave Erev rows
  __shared__ float QEs[4][16][32];               // per-wave Q@Erev^T tiles
  const int t = threadIdx.x, lane = t & 31, w = t >> 5;
  const int bh = blockIdx.y;
  const int i0 = blockIdx.x * 16;
  // Q tile 16x64 bf16: async DMA copy (2 KB, 16 B per thread)
  async_b128(Qh + ((size_t)bh * Ln + i0) * DHn + t * 8, (u16*)Qs + t * 8);
  const float scale = 0.125f;                    // 1/sqrt(64)
  for (int j0 = 0; j0 < Ln; j0 += 64) {
    { // K tile 64x64 bf16 (8 KB): TDM if available, else async copies
      const u16* src = Kh + ((size_t)bh * Ln + j0) * DHn;
#if HAVE_TDM
      if (w == 0) {
        tdm_load_tile_2d_b16((u32)(size_t)&Ks[0][0], src, 64u, 64u, 64u);
        __builtin_amdgcn_s_wait_tensorcnt(0);
      }
#else
      #pragma unroll
      for (int p = 0; p < 4; ++p) {
        int idx = t + 128 * p;
        async_b128(src + idx * 8, (u16*)Ks + idx * 8);
      }
#endif
    }
    const int jw = j0 + w * 16;
    const int rbase = i0 - jw - 15;              // min relative offset this tile
    { // Erev rows rbase..rbase+31: Erev[r] = E[M-1-r], clamped (masked later)
      int r = rbase + lane;
      int erow = Mseq - 1 - r;
      erow = erow < 0 ? 0 : (erow > Mseq - 1 ? Mseq - 1 : erow);
      const f32x4* src = (const f32x4*)(E + (size_t)erow * DHn);
      #pragma unroll
      for (int c = 0; c < 16; ++c) {
        f32x4 v = src[c];
        uint2 pv = { pk2(v[0], v[1]), pk2(v[2], v[3]) };
        *(uint2*)&Es[w][lane][c * 4] = pv;
      }
    }
    async_wait();
    __syncthreads();
    bf16x16 aq0 = load_frag(&Qs[0][0], 64);
    bf16x16 aq1 = load_frag(&Qs[0][32], 64);
    f32x8 s = {};
    s = wmma_bf16(aq0, load_frag(&Ks[w * 16][0], 64), s);
    s = wmma_bf16(aq1, load_frag(&Ks[w * 16][32], 64), s);
    #pragma unroll
    for (int rt = 0; rt < 2; ++rt) {             // two 16x16 Q@Erev^T tiles
      f32x8 qe = {};
      qe = wmma_bf16(aq0, load_frag(&Es[w][rt * 16][0], 64), qe);
      qe = wmma_bf16(aq1, load_frag(&Es[w][rt * 16][32], 64), qe);
      #pragma unroll
      for (int e = 0; e < 8; ++e)
        QEs[w][e + ((lane & 16) >> 1)][rt * 16 + (lane & 15)] = qe[e];
    }
    // same-wave LDS ops are in-order: gather anti-diagonal, add, scale, store
    const int hi8 = (lane & 16) >> 1;
    #pragma unroll
    for (int e = 0; e < 8; ++e) {
      int il = e + hi8;
      int i = i0 + il;
      int n = lane & 15;
      int j = jw + n;
      float srel = (i - j >= 0) ? QEs[w][il][il + 15 - n] : 0.0f;
      attn[((size_t)bh * Ln + i) * Ln + j] = (s[e] + srel) * scale;
    }
    __syncthreads();
  }
}

// ---------------- Kernel 3: row softmax in place (one block per row)
__global__ __launch_bounds__(256)
void softmax_kernel(float* __restrict__ attn) {
  float* p = attn + (size_t)blockIdx.x * Ln;
  __shared__ float red[256];
  const int t = threadIdx.x;
  float v[8], mx = -1e30f;
  #pragma unroll
  for (int i = 0; i < 8; ++i) { v[i] = p[t + 256 * i]; mx = fmaxf(mx, v[i]); }
  red[t] = mx; __syncthreads();
  for (int s = 128; s > 0; s >>= 1) {
    if (t < s) red[t] = fmaxf(red[t], red[t + s]);
    __syncthreads();
  }
  mx = red[0]; __syncthreads();
  float sum = 0.f;
  #pragma unroll
  for (int i = 0; i < 8; ++i) { v[i] = __expf(v[i] - mx); sum += v[i]; }
  red[t] = sum; __syncthreads();
  for (int s = 128; s > 0; s >>= 1) {
    if (t < s) red[t] += red[t + s];
    __syncthreads();
  }
  float inv = 1.0f / red[0];
  #pragma unroll
  for (int i = 0; i < 8; ++i) p[t + 256 * i] = v[i] * inv;
}

// ---------------- Kernel 4: ctx = attn_w @ V  (per b,h) -> bf16 (B*L, D)
__global__ __launch_bounds__(128)
void av_gemm_kernel(const float* __restrict__ attn, const u16* __restrict__ Vh,
                    u16* __restrict__ ctx) {
  __shared__ __align__(16) u16 As[64][32];
  __shared__ __align__(16) u16 Bs[64][32];
  const int t = threadIdx.x, lane = t & 31, w = t >> 5;
  const int bh = blockIdx.y, b = bh >> 3, h = bh & 7;
  const int i0 = blockIdx.x * 64;
  const float* Ab = attn + (size_t)bh * Ln * Ln;
  const u16*   Vb = Vh   + (size_t)bh * Ln * DHn;
  f32x8 acc[4] = {};
  for (int k0 = 0; k0 < Ln; k0 += 32) {
    #pragma unroll
    for (int p = 0; p < 4; ++p) {           // attn f32 -> packed bf16 tile
      int row = p * 16 + (t >> 3);
      int col = (t & 7) * 4;
      f32x4 v = *(const f32x4*)(Ab + (size_t)(i0 + row) * Ln + k0 + col);
      uint2 pv = { pk2(v[0], v[1]), pk2(v[2], v[3]) };
      *(uint2*)&As[row][col] = pv;
    }
    {                                       // V bf16, column-per-thread transpose
      int dh = t & 63;
      int kg = (t >> 6) * 16;
      union { uint4 u4[2]; u16 us[16]; } buf;
      #pragma unroll
      for (int kk = 0; kk < 16; ++kk)
        buf.us[kk] = Vb[(size_t)(k0 + kg + kk) * DHn + dh];
      *(uint4*)&Bs[dh][kg]     = buf.u4[0];
      *(uint4*)&Bs[dh][kg + 8] = buf.u4[1];
    }
    if (k0 + 32 < Ln)
      __builtin_prefetch(Ab + (size_t)(i0 + (t >> 1)) * Ln + k0 + 32, 0, 0);
    __syncthreads();
    bf16x16 a = load_frag(&As[w * 16][0], 32);
    #pragma unroll
    for (int ns = 0; ns < 4; ++ns)
      acc[ns] = wmma_bf16(a, load_frag(&Bs[ns * 16][0], 32), acc[ns]);
    __syncthreads();
  }
  const int hi8 = (lane & 16) >> 1;
  #pragma unroll
  for (int ns = 0; ns < 4; ++ns) {
    int dh = ns * 16 + (lane & 15);
    #pragma unroll
    for (int e = 0; e < 8; ++e) {
      int i = i0 + w * 16 + e + hi8;
      ctx[((size_t)(b * Ln + i)) * Dn + h * DHn + dh] = f2bf(acc[ns][e]);
    }
  }
}

// ---------------- Kernel 5: out = ctx @ fc_w + fc_b  (f32 out)
__global__ __launch_bounds__(128)
void fc_gemm_kernel(const u16* __restrict__ Xh, const float* __restrict__ W,
                    const float* __restrict__ bias, float* __restrict__ out) {
  __shared__ __align__(16) u16 As[64][32];
  __shared__ __align__(16) u16 Bs[64][32];
  const int t = threadIdx.x, lane = t & 31, w = t >> 5;
  const int n0 = blockIdx.x * 64;
  const int m0 = blockIdx.y * 64;
  f32x8 acc[4] = {};
  for (int k0 = 0; k0 < Dn; k0 += 32) {
    #pragma unroll
    for (int p = 0; p < 2; ++p) {           // A already bf16: async DMA copy
      int idx = p * 128 + t;                // 0..255, 16 B each
      int row = idx >> 2, cg = idx & 3;
      async_b128(Xh + (size_t)(m0 + row) * Dn + k0 + cg * 8, &As[row][cg * 8]);
    }
    {                                       // W: column-per-thread, packed store
      int n  = t & 63;
      int kg = (t >> 6) * 16;
      #pragma unroll
      for (int kk = 0; kk < 16; kk += 2) {
        float a = W[(size_t)(k0 + kg + kk)     * Dn + n0 + n];
        float b = W[(size_t)(k0 + kg + kk + 1) * Dn + n0 + n];
        *(u32*)&Bs[n][kg + kk] = pk2(a, b);
      }
    }
    if (k0 + 32 < Dn)
      __builtin_prefetch(W + (size_t)(k0 + 32 + (t >> 2)) * Dn + n0, 0, 0);
    async_wait();
    __syncthreads();
    bf16x16 a = load_frag(&As[w * 16][0], 32);
    #pragma unroll
    for (int ns = 0; ns < 4; ++ns)
      acc[ns] = wmma_bf16(a, load_frag(&Bs[ns * 16][0], 32), acc[ns]);
    __syncthreads();
  }
  const int hi8 = (lane & 16) >> 1;
  #pragma unroll
  for (int ns = 0; ns < 4; ++ns) {
    int n = n0 + ns * 16 + (lane & 15);
    float bv = bias[n];
    #pragma unroll
    for (int e = 0; e < 8; ++e) {
      int m = m0 + w * 16 + e + hi8;
      out[(size_t)m * Dn + n] = acc[ns][e] + bv;
    }
  }
}

extern "C" void kernel_launch(void* const* d_in, const int* in_sizes, int n_in,
                              void* d_out, int out_size, void* d_ws, size_t ws_size,
                              hipStream_t stream) {
  const float* q_in = (const float*)d_in[0];
  const float* k_in = (const float*)d_in[1];
  const float* v_in = (const float*)d_in[2];
  const float* Wq_w = (const float*)d_in[3];
  const float* Wq_b = (const float*)d_in[4];
  const float* Wk_w = (const float*)d_in[5];
  const float* Wk_b = (const float*)d_in[6];
  const float* Wv_w = (const float*)d_in[7];
  const float* Wv_b = (const float*)d_in[8];
  const float* E    = (const float*)d_in[9];
  const float* fc_w = (const float*)d_in[10];
  const float* fc_b = (const float*)d_in[11];

  float* out  = (float*)d_out;                       // (B, L, D)
  float* attn = out + (size_t)Bn * Ln * Dn;          // (B, H, L, L)

  const size_t per = (size_t)Bn * Hn * Ln * DHn;     // 4,194,304 elems
  u16* qh  = (u16*)d_ws;
  u16* kh  = qh + per;
  u16* vh  = kh + per;
  u16* ctx = vh + per;                               // (B*L, D) bf16

  dim3 gProj(Dn / 64, (Bn * Ln) / 64);               // (8, 128)
  proj_gemm_kernel<<<gProj, 128, 0, stream>>>(q_in, Wq_w, Wq_b, qh);
  proj_gemm_kernel<<<gProj, 128, 0, stream>>>(k_in, Wk_w, Wk_b, kh);
  proj_gemm_kernel<<<gProj, 128, 0, stream>>>(v_in, Wv_w, Wv_b, vh);

  logits_kernel<<<dim3(Ln / 16, Bn * Hn), 128, 0, stream>>>(qh, kh, E, attn);
  softmax_kernel<<<dim3(Bn * Hn * Ln), 256, 0, stream>>>(attn);
  av_gemm_kernel<<<dim3(Ln / 64, Bn * Hn), 128, 0, stream>>>(attn, vh, ctx);
  fc_gemm_kernel<<<dim3(Dn / 64, (Bn * Ln) / 64), 128, 0, stream>>>(ctx, fc_w, fc_b, out);
}
